// XciT_16466904613641
// MI455X (gfx1250) — compile-verified, matrix operand order checked by reading, not run on hardware
//
#include <hip/hip_runtime.h>
#include <hip/hip_bf16.h>
#include <math.h>

typedef __attribute__((ext_vector_type(16))) _Float16 v16h;
typedef __attribute__((ext_vector_type(8)))  _Float16 v8h;
typedef __attribute__((ext_vector_type(8)))  float    v8f;
typedef __attribute__((ext_vector_type(4)))  float    v4f;

__device__ __forceinline__ float gelu_f(float x) {
    return 0.5f * x * (1.0f + erff(x * 0.70710678118654752f));
}

// ---------------------------------------------------------------------------
// Tiled WMMA GEMM:  C[M,N] = act(A[M,K] * W[N,K]^T + bias) * scale + resid
// Block: 256 threads (8 waves, wave32). Tile 128(M) x 64(N), K-step 32.
// Double-buffered LDS (ping-pong), vectorized float4 global loads with
// row-level predication, packed b128 LDS stores, fragments gathered per the
// CDNA5 16-bit WMMA VGPR layout (ISA 7.12.2).
// ---------------------------------------------------------------------------
__global__ __launch_bounds__(256)
void gemm_kernel(const float* __restrict__ A, int lda,
                 const float* __restrict__ W,
                 const float* __restrict__ bias,
                 float* __restrict__ C, int ldc,
                 const float* __restrict__ scale,
                 const float* __restrict__ resid, int ldres,
                 int M, int N, int K, int act)
{
    __shared__ _Float16 As[2][128][40];  // 32 K-halves + pad, x2 buffers
    __shared__ _Float16 Bs[2][64][40];

    const int tid  = threadIdx.x;
    const int bm   = blockIdx.y * 128;
    const int bn   = blockIdx.x * 64;
    const int wv   = tid >> 5;
    const int lane = tid & 31;
    const int wm   = wv >> 1;     // 0..3 -> 32-row band
    const int wn   = wv & 1;      // 0..1 -> 32-col band
    const int lhi  = lane >> 4;   // 0/1
    const int llo  = lane & 15;

    // staging assignment
    const int ar   = tid >> 1;            // A row 0..127
    const int aseg = (tid & 1) * 16;      // A col segment
    const int br   = tid >> 2;            // B row 0..63
    const int bseg = (tid & 3) * 8;       // B col segment
    const int gmA  = bm + ar;
    const int gnB  = bn + br;
    const bool aok = gmA < M;
    const bool bok = gnB < N;
    const float* aBase = A + (long long)gmA * lda + aseg;
    const float* bBase = W + (long long)gnB * K + bseg;

    v4f aR[4], bR[2];

    auto loadA = [&](int k0) {
        if (aok) {
            const v4f* p = reinterpret_cast<const v4f*>(aBase + k0);
            aR[0] = p[0]; aR[1] = p[1]; aR[2] = p[2]; aR[3] = p[3];
        } else {
            v4f z = {0.0f, 0.0f, 0.0f, 0.0f};
            aR[0] = z; aR[1] = z; aR[2] = z; aR[3] = z;
        }
    };
    auto loadB = [&](int k0) {
        if (bok) {
            const v4f* p = reinterpret_cast<const v4f*>(bBase + k0);
            bR[0] = p[0]; bR[1] = p[1];
        } else {
            v4f z = {0.0f, 0.0f, 0.0f, 0.0f};
            bR[0] = z; bR[1] = z;
        }
    };
    auto storeA = [&](int buf) {
        v8h h0, h1;
        #pragma unroll
        for (int i = 0; i < 4; ++i) { h0[i] = (_Float16)aR[0][i]; h0[4 + i] = (_Float16)aR[1][i]; }
        #pragma unroll
        for (int i = 0; i < 4; ++i) { h1[i] = (_Float16)aR[2][i]; h1[4 + i] = (_Float16)aR[3][i]; }
        *reinterpret_cast<v8h*>(&As[buf][ar][aseg])     = h0;
        *reinterpret_cast<v8h*>(&As[buf][ar][aseg + 8]) = h1;
    };
    auto storeB = [&](int buf) {
        v8h h0;
        #pragma unroll
        for (int i = 0; i < 4; ++i) { h0[i] = (_Float16)bR[0][i]; h0[4 + i] = (_Float16)bR[1][i]; }
        *reinterpret_cast<v8h*>(&Bs[buf][br][bseg]) = h0;
    };

    v8f acc[2][2];
    #pragma unroll
    for (int i = 0; i < 2; ++i)
        #pragma unroll
        for (int j = 0; j < 2; ++j)
            #pragma unroll
            for (int r = 0; r < 8; ++r) acc[i][j][r] = 0.0f;

    const int nk = K >> 5;

    // prologue: stage tile 0 into buffer 0
    loadA(0); loadB(0);
    storeA(0); storeB(0);
    __syncthreads();

    for (int kt = 0; kt < nk; ++kt) {
        const int cur = kt & 1;
        const bool more = (kt + 1) < nk;
        if (more) { loadA((kt + 1) << 5); loadB((kt + 1) << 5); }

        // gather fragments from LDS (CDNA5 16-bit A/B layout) and issue WMMAs
        v16h af[2], bf[2];
        #pragma unroll
        for (int f = 0; f < 2; ++f) {
            int m = wm * 32 + f * 16 + llo;
            #pragma unroll
            for (int j = 0; j < 8; ++j) {
                int kk = 16 * (j >> 2) + 8 * lhi + 2 * (j & 3);
                af[f][2 * j]     = As[cur][m][kk];
                af[f][2 * j + 1] = As[cur][m][kk + 1];
            }
        }
        #pragma unroll
        for (int f = 0; f < 2; ++f) {
            int n = wn * 32 + f * 16 + llo;
            #pragma unroll
            for (int j = 0; j < 8; ++j) {
                int kk = 16 * (j >> 2) + 8 * lhi + 2 * (j & 3);
                bf[f][2 * j]     = Bs[cur][n][kk];
                bf[f][2 * j + 1] = Bs[cur][n][kk + 1];
            }
        }
        #pragma unroll
        for (int i = 0; i < 2; ++i)
            #pragma unroll
            for (int j = 0; j < 2; ++j)
                acc[i][j] = __builtin_amdgcn_wmma_f32_16x16x32_f16(
                    false, af[i], false, bf[j], (short)0, acc[i][j], false, false);

        if (more) { storeA(1 - cur); storeB(1 - cur); }
        __syncthreads();
    }

    // Epilogue: D layout -> lane holds col llo, rows r + 8*lhi
    #pragma unroll
    for (int i = 0; i < 2; ++i) {
        #pragma unroll
        for (int j = 0; j < 2; ++j) {
            int gn = bn + wn * 32 + j * 16 + llo;
            #pragma unroll
            for (int r = 0; r < 8; ++r) {
                int gm = bm + wm * 32 + i * 16 + r + 8 * lhi;
                if (gm < M && gn < N) {
                    float v = acc[i][j][r];
                    if (bias)  v += bias[gn];
                    if (act == 1) v = gelu_f(v);
                    if (scale) v *= scale[gn];
                    if (resid) v += resid[(long long)gm * ldres + gn];
                    C[(long long)gm * ldc + gn] = v;
                }
            }
        }
    }
}

// ---------------------------------------------------------------------------
// LayerNorm over D=384, one block (192 threads) per row.
// ---------------------------------------------------------------------------
__global__ __launch_bounds__(192)
void ln_kernel(const float* __restrict__ X, long long xstride,
               float* __restrict__ Y, long long ystride,
               const float* __restrict__ w, const float* __restrict__ b)
{
    __shared__ float s1[192], s2[192];
    int row = blockIdx.x;
    int t = threadIdx.x;
    const float* x = X + (long long)row * xstride;
    float a0 = x[t], a1 = x[t + 192];
    s1[t] = a0 + a1;
    s2[t] = a0 * a0 + a1 * a1;
    __syncthreads();
    for (int off = 96; off >= 6; off >>= 1) {
        if (t < off) { s1[t] += s1[t + off]; s2[t] += s2[t + off]; }
        __syncthreads();
    }
    float sum = s1[0] + s1[1] + s1[2] + s1[3] + s1[4] + s1[5];
    float sq  = s2[0] + s2[1] + s2[2] + s2[3] + s2[4] + s2[5];
    float mean = sum * (1.0f / 384.0f);
    float var  = sq  * (1.0f / 384.0f) - mean * mean;
    float inv  = rsqrtf(var + 1e-5f);
    float* y = Y + (long long)row * ystride;
    y[t]       = (a0 - mean) * inv * w[t]       + b[t];
    y[t + 192] = (a1 - mean) * inv * w[t + 192] + b[t + 192];
}

// ---------------------------------------------------------------------------
// Direct 3x3 stride-2 pad-1 conv with fused eval-BN (+optional exact GELU).
// ---------------------------------------------------------------------------
__global__ void conv3x3s2_kernel(const float* __restrict__ x, const float* __restrict__ w,
                                 const float* __restrict__ g, const float* __restrict__ bb,
                                 float* __restrict__ y,
                                 int Bn, int Ci, int Hi, int Wi, int Co, int dogelu)
{
    int Ho = Hi >> 1, Wo = Wi >> 1;
    long long total = (long long)Bn * Co * Ho * Wo;
    long long idx = (long long)blockIdx.x * blockDim.x + threadIdx.x;
    if (idx >= total) return;
    int wo = (int)(idx % Wo);
    int ho = (int)((idx / Wo) % Ho);
    int co = (int)((idx / ((long long)Wo * Ho)) % Co);
    int b  = (int)(idx / ((long long)Wo * Ho * Co));
    float acc = 0.0f;
    const float* wp = w + (long long)co * Ci * 9;
    for (int ci = 0; ci < Ci; ++ci) {
        const float* xp = x + ((long long)(b * Ci + ci) * Hi) * Wi;
        #pragma unroll
        for (int ky = 0; ky < 3; ++ky) {
            int yi = ho * 2 - 1 + ky;
            if (yi < 0 || yi >= Hi) continue;
            #pragma unroll
            for (int kx = 0; kx < 3; ++kx) {
                int xi = wo * 2 - 1 + kx;
                if (xi < 0 || xi >= Wi) continue;
                acc += xp[(long long)yi * Wi + xi] * wp[ci * 9 + ky * 3 + kx];
            }
        }
    }
    acc = acc * (g[co] * rsqrtf(1.0f + 1e-5f)) + bb[co];
    if (dogelu) acc = gelu_f(acc);
    y[idx] = acc;
}

// ---------------------------------------------------------------------------
// Sine/cosine positional basis [196,64] and projection to [196,384].
// ---------------------------------------------------------------------------
__global__ void basis_kernel(float* __restrict__ basis)
{
    int idx = blockIdx.x * blockDim.x + threadIdx.x;
    if (idx >= 196 * 64) return;
    int n = idx / 64, c = idx % 64;
    int hh = n / 14, ww = n % 14;
    const float two_pi = 6.283185307179586f;
    float embed;
    int cc = c;
    if (c < 32) {
        embed = (hh + 1.0f) / ((14.0f + 1e-6f) * two_pi);
    } else {
        embed = (ww + 1.0f) / ((14.0f + 1e-6f) * two_pi);
        cc = c - 32;
    }
    int p = cc >> 1;
    float dimt = powf(10000.0f, (2.0f * p) / 32.0f);
    float arg = embed / dimt;
    basis[idx] = (cc & 1) ? cosf(arg) : sinf(arg);
}

__global__ void pos_kernel(const float* __restrict__ basis, const float* __restrict__ pw,
                           const float* __restrict__ pb, float* __restrict__ pos)
{
    int idx = blockIdx.x * blockDim.x + threadIdx.x;
    if (idx >= 196 * 384) return;
    int n = idx / 384, d = idx % 384;
    float s = 0.0f;
    #pragma unroll 8
    for (int c = 0; c < 64; ++c) s += basis[n * 64 + c] * pw[d * 64 + c];
    pos[idx] = s + pb[d];
}

__global__ void build_t_kernel(const float* __restrict__ c4, const float* __restrict__ pos,
                               float* __restrict__ t)
{
    int idx = blockIdx.x * blockDim.x + threadIdx.x;
    if (idx >= 32 * 196 * 384) return;
    int d = idx % 384;
    int n = (idx / 384) % 196;
    int b = idx / (384 * 196);
    t[idx] = c4[((long long)(b * 384 + d)) * 196 + n] + pos[n * 384 + d];
}

// ---------------------------------------------------------------------------
// XCA cross-covariance attention, one workgroup per (b,h).
// q,k,v from qkv[B,N,3,NH,HD]; normalize over N; 48x48 attn; softmax; a*v.
// ---------------------------------------------------------------------------
__global__ __launch_bounds__(256)
void xca_attn_kernel(const float* __restrict__ qkv, const float* __restrict__ temp,
                     float* __restrict__ o)
{
    __shared__ _Float16 qh[48][200];
    __shared__ _Float16 kh[48][200];
    __shared__ float att[48][48];
    __shared__ float invn[96];
    int bh = blockIdx.x;
    int b = bh >> 3, h = bh & 7;
    int tid = threadIdx.x;
    const float* base = qkv + (long long)b * 196 * 1152 + h * 48;
    for (int idx = tid; idx < 48 * 196; idx += 256) {
        int d = idx % 48, n = idx / 48;
        qh[d][n] = (_Float16)base[(long long)n * 1152 + d];
        kh[d][n] = (_Float16)base[(long long)n * 1152 + 384 + d];
    }
    __syncthreads();
    if (tid < 96) {
        int d = tid % 48;
        bool isk = tid >= 48;
        float s = 0.0f;
        for (int n = 0; n < 196; ++n) {
            float v = isk ? (float)kh[d][n] : (float)qh[d][n];
            s += v * v;
        }
        invn[tid] = 1.0f / fmaxf(sqrtf(s), 1e-12f);
    }
    __syncthreads();
    float th = temp[h];
    for (int idx = tid; idx < 48 * 48; idx += 256) {
        int d = idx / 48, e = idx % 48;
        float s = 0.0f;
        for (int n = 0; n < 196; ++n) s += (float)qh[d][n] * (float)kh[e][n];
        att[d][e] = s * invn[d] * invn[48 + e] * th;
    }
    __syncthreads();
    if (tid < 48) {
        int d = tid;
        float m = -1e30f;
        for (int e = 0; e < 48; ++e) m = fmaxf(m, att[d][e]);
        float s = 0.0f;
        for (int e = 0; e < 48; ++e) { float ex = expf(att[d][e] - m); att[d][e] = ex; s += ex; }
        float r = 1.0f / s;
        for (int e = 0; e < 48; ++e) att[d][e] *= r;
    }
    __syncthreads();
    for (int idx = tid; idx < 48 * 196; idx += 256) {  // reuse qh for v
        int d = idx % 48, n = idx / 48;
        qh[d][n] = (_Float16)base[(long long)n * 1152 + 768 + d];
    }
    __syncthreads();
    float* ob = o + (long long)b * 196 * 384 + h * 48;
    for (int idx = tid; idx < 48 * 196; idx += 256) {
        int d = idx % 48, n = idx / 48;
        float s = 0.0f;
        for (int e = 0; e < 48; ++e) s += att[d][e] * (float)qh[e][n];
        ob[(long long)n * 384 + d] = s;
    }
}

// ---------------------------------------------------------------------------
// LPI block: depthwise 3x3 -> GELU -> BN -> depthwise 3x3; t += g3 * out.
// One workgroup per (b,d); 14x14 image (+1 halo) in LDS.
// ---------------------------------------------------------------------------
__global__ __launch_bounds__(256)
void lpi_kernel(const float* __restrict__ hbuf,
                const float* __restrict__ w1, const float* __restrict__ b1,
                const float* __restrict__ bng, const float* __restrict__ bnb,
                const float* __restrict__ w2, const float* __restrict__ b2,
                const float* __restrict__ g3, float* __restrict__ t)
{
    __shared__ float img[16][16];
    __shared__ float tmp[16][16];
    int bd = blockIdx.x;
    int b = bd / 384, d = bd % 384;
    int tid = threadIdx.x;
    img[tid >> 4][tid & 15] = 0.0f;
    tmp[tid >> 4][tid & 15] = 0.0f;
    __syncthreads();
    float wl1[9], wl2[9];
    #pragma unroll
    for (int i = 0; i < 9; ++i) { wl1[i] = w1[d * 9 + i]; wl2[i] = w2[d * 9 + i]; }
    int n = tid;
    if (n < 196) {
        int y = n / 14, x = n % 14;
        img[y + 1][x + 1] = hbuf[((long long)b * 196 + n) * 384 + d];
    }
    __syncthreads();
    float bnscale = bng[d] * rsqrtf(1.0f + 1e-5f);
    if (n < 196) {
        int y = n / 14, x = n % 14;
        float s = b1[d];
        #pragma unroll
        for (int ky = 0; ky < 3; ++ky)
            #pragma unroll
            for (int kx = 0; kx < 3; ++kx)
                s += img[y + ky][x + kx] * wl1[ky * 3 + kx];
        s = gelu_f(s);
        tmp[y + 1][x + 1] = s * bnscale + bnb[d];
    }
    __syncthreads();
    if (n < 196) {
        int y = n / 14, x = n % 14;
        float s = b2[d];
        #pragma unroll
        for (int ky = 0; ky < 3; ++ky)
            #pragma unroll
            for (int kx = 0; kx < 3; ++kx)
                s += tmp[y + ky][x + kx] * wl2[ky * 3 + kx];
        t[((long long)b * 196 + n) * 384 + d] += g3[d] * s;
    }
}

// ---------------------------------------------------------------------------
// Class-attention + helpers for the two CA blocks.
// ---------------------------------------------------------------------------
__global__ void build_t2_kernel(const float* __restrict__ t, const float* __restrict__ cls,
                                float* __restrict__ t2)
{
    int idx = blockIdx.x * blockDim.x + threadIdx.x;
    if (idx >= 32 * 197 * 384) return;
    int d = idx % 384;
    int n = (idx / 384) % 197;
    int b = idx / (384 * 197);
    t2[idx] = (n == 0) ? cls[d] : t[((long long)b * 196 + (n - 1)) * 384 + d];
}

__global__ __launch_bounds__(256)
void ca_attn_kernel(const float* __restrict__ qkv, float* __restrict__ clsb)
{
    __shared__ float q1[48];
    __shared__ float att[197];
    __shared__ float red[1];
    int bh = blockIdx.x;
    int b = bh >> 3, h = bh & 7;
    int tid = threadIdx.x;
    const float* base = qkv + (long long)b * 197 * 1152 + h * 48;
    if (tid < 48) q1[tid] = base[tid];  // q of token 0
    __syncthreads();
    if (tid < 197) {
        const float* kp = base + (long long)tid * 1152 + 384;
        float s = 0.0f;
        #pragma unroll 8
        for (int d = 0; d < 48; ++d) s += q1[d] * kp[d];
        att[tid] = s * 0.14433756729740643f;  // 48^-0.5
    }
    __syncthreads();
    if (tid == 0) {
        float m = -1e30f;
        for (int n = 0; n < 197; ++n) m = fmaxf(m, att[n]);
        float s = 0.0f;
        for (int n = 0; n < 197; ++n) { float e = expf(att[n] - m); att[n] = e; s += e; }
        red[0] = 1.0f / s;
    }
    __syncthreads();
    float rs = red[0];
    if (tid < 48) {
        float s = 0.0f;
        for (int n = 0; n < 197; ++n)
            s += att[n] * rs * base[(long long)n * 1152 + 768 + tid];
        clsb[b * 384 + h * 48 + tid] = s;
    }
}

__global__ void ca_merge1_kernel(float* __restrict__ t2, const float* __restrict__ h,
                                 const float* __restrict__ clsp, const float* __restrict__ g1)
{
    int idx = blockIdx.x * blockDim.x + threadIdx.x;
    if (idx >= 32 * 197 * 384) return;
    int d = idx % 384;
    int n = (idx / 384) % 197;
    int b = idx / (384 * 197);
    float add = (n == 0) ? clsp[b * 384 + d] : h[idx];
    t2[idx] += g1[d] * add;
}

__global__ void ca_merge2_kernel(float* __restrict__ t2, const float* __restrict__ cls2,
                                 const float* __restrict__ g2)
{
    int idx = blockIdx.x * blockDim.x + threadIdx.x;
    if (idx >= 32 * 197 * 384) return;
    int d = idx % 384;
    int n = (idx / 384) % 197;
    int b = idx / (384 * 197);
    if (n == 0) t2[idx] += g2[d] * cls2[b * 384 + d];
    else        t2[idx] *= 2.0f;
}

// ---------------------------------------------------------------------------
// Host orchestration
// ---------------------------------------------------------------------------
static void gemm(const float* A, int lda, const float* W, const float* bias,
                 float* C, int ldc, const float* scale, const float* resid, int ldres,
                 int M, int N, int K, int act, hipStream_t s)
{
    dim3 g((N + 63) / 64, (M + 127) / 128);
    gemm_kernel<<<g, 256, 0, s>>>(A, lda, W, bias, C, ldc, scale, resid, ldres, M, N, K, act);
}

extern "C" void kernel_launch(void* const* d_in, const int* in_sizes, int n_in,
                              void* d_out, int out_size, void* d_ws, size_t ws_size,
                              hipStream_t stream)
{
    (void)in_sizes; (void)n_in; (void)out_size; (void)ws_size;
    const float* x         = (const float*)d_in[0];
    const float* cls_token = (const float*)d_in[1];
    const float* pos_w     = (const float*)d_in[2];
    const float* pos_b     = (const float*)d_in[3];
    const float* pe_w[4] = {(const float*)d_in[4], (const float*)d_in[7], (const float*)d_in[10], (const float*)d_in[13]};
    const float* pe_g[4] = {(const float*)d_in[5], (const float*)d_in[8], (const float*)d_in[11], (const float*)d_in[14]};
    const float* pe_b[4] = {(const float*)d_in[6], (const float*)d_in[9], (const float*)d_in[12], (const float*)d_in[15]};
    const float* xqkv_w  = (const float*)d_in[16];
    const float* xqkv_b  = (const float*)d_in[17];
    const float* xproj_w = (const float*)d_in[18];
    const float* xproj_b = (const float*)d_in[19];
    const float* xtemp   = (const float*)d_in[20];
    const float* xn1_w = (const float*)d_in[21]; const float* xn1_b = (const float*)d_in[22];
    const float* xn2_w = (const float*)d_in[23]; const float* xn2_b = (const float*)d_in[24];
    const float* xn3_w = (const float*)d_in[25]; const float* xn3_b = (const float*)d_in[26];
    const float* xfc1_w = (const float*)d_in[27]; const float* xfc1_b = (const float*)d_in[28];
    const float* xfc2_w = (const float*)d_in[29]; const float* xfc2_b = (const float*)d_in[30];
    const float* xlpi1_w = (const float*)d_in[31]; const float* xlpi1_b = (const float*)d_in[32];
    const float* xlpi2_w = (const float*)d_in[33]; const float* xlpi2_b = (const float*)d_in[34];
    const float* xbn_g = (const float*)d_in[35]; const float* xbn_b = (const float*)d_in[36];
    const float* xg1 = (const float*)d_in[37];
    const float* xg2 = (const float*)d_in[38];
    const float* xg3 = (const float*)d_in[39];
    const float* cqkv_w = (const float*)d_in[40]; const float* cqkv_b = (const float*)d_in[41];
    const float* cproj_w = (const float*)d_in[42]; const float* cproj_b = (const float*)d_in[43];
    const float* cn1_w = (const float*)d_in[44]; const float* cn1_b = (const float*)d_in[45];
    const float* cn2_w = (const float*)d_in[46]; const float* cn2_b = (const float*)d_in[47];
    const float* cfc1_w = (const float*)d_in[48]; const float* cfc1_b = (const float*)d_in[49];
    const float* cfc2_w = (const float*)d_in[50]; const float* cfc2_b = (const float*)d_in[51];
    const float* cg1 = (const float*)d_in[52]; const float* cg2 = (const float*)d_in[53];
    const float* norm_w = (const float*)d_in[54]; const float* norm_b = (const float*)d_in[55];
    const float* head_w = (const float*)d_in[56]; const float* head_b = (const float*)d_in[57];

    float* ws = (float*)d_ws;

    // Workspace layout (floats). Conv ping-pong reuses transformer regions.
    const long long OFF_T     = 0;                       // [6272,384]
    const long long OFF_H     = OFF_T   + 6272LL * 384;  // [6304,384]
    const long long OFF_O     = OFF_H   + 6304LL * 384;  // [6272,384]
    const long long OFF_POS   = OFF_O   + 6272LL * 384;  // [196,384]
    const long long OFF_BASIS = OFF_POS + 196LL * 384;   // [196,64]
    const long long OFF_QKV   = OFF_BASIS + 196LL * 64;  // [6304,1152]
    const long long OFF_HID   = OFF_QKV + 6304LL * 1152; // [6304,1536]
    const long long OFF_T2    = OFF_HID + 6304LL * 1536; // [6304,384]
    const long long OFF_CLSB  = OFF_T2  + 6304LL * 384;  // [32,384]
    const long long OFF_CLSP  = OFF_CLSB + 32LL * 384;
    const long long OFF_CLS2  = OFF_CLSP + 32LL * 384;
    const long long OFF_CLSH  = OFF_CLS2 + 32LL * 384;   // [32,1536]

    // ---- patch embed convs (ping-pong) ----
    float* c1 = ws + 0;
    float* c2 = ws + 19267584LL;  // 32*48*112*112
    float* c3 = ws + 0;
    float* c4 = ws + 19267584LL;
    {
        long long tot = 32LL * 48 * 112 * 112;
        conv3x3s2_kernel<<<(int)((tot + 255) / 256), 256, 0, stream>>>(x,  pe_w[0], pe_g[0], pe_b[0], c1, 32, 3,   224, 224, 48,  1);
        tot = 32LL * 96 * 56 * 56;
        conv3x3s2_kernel<<<(int)((tot + 255) / 256), 256, 0, stream>>>(c1, pe_w[1], pe_g[1], pe_b[1], c2, 32, 48,  112, 112, 96,  1);
        tot = 32LL * 192 * 28 * 28;
        conv3x3s2_kernel<<<(int)((tot + 255) / 256), 256, 0, stream>>>(c2, pe_w[2], pe_g[2], pe_b[2], c3, 32, 96,  56,  56,  192, 1);
        tot = 32LL * 384 * 14 * 14;
        conv3x3s2_kernel<<<(int)((tot + 255) / 256), 256, 0, stream>>>(c3, pe_w[3], pe_g[3], pe_b[3], c4, 32, 192, 28,  28,  384, 0);
    }

    // ---- positional embedding + token build ----
    basis_kernel<<<(196 * 64 + 255) / 256, 256, 0, stream>>>(ws + OFF_BASIS);
    pos_kernel<<<(196 * 384 + 255) / 256, 256, 0, stream>>>(ws + OFF_BASIS, pos_w, pos_b, ws + OFF_POS);
    build_t_kernel<<<(32 * 196 * 384 + 255) / 256, 256, 0, stream>>>(c4, ws + OFF_POS, ws + OFF_T);

    float* tb   = ws + OFF_T;
    float* hb   = ws + OFF_H;
    float* ob   = ws + OFF_O;
    float* qkvb = ws + OFF_QKV;
    float* hidb = ws + OFF_HID;

    // ---- 24 XCA layers ----
    for (int l = 0; l < 24; ++l) {
        ln_kernel<<<6272, 192, 0, stream>>>(tb, 384, hb, 384, xn1_w + l * 384, xn1_b + l * 384);
        gemm(hb, 384, xqkv_w + (long long)l * 1152 * 384, xqkv_b + l * 1152,
             qkvb, 1152, nullptr, nullptr, 0, 6272, 1152, 384, 0, stream);
        xca_attn_kernel<<<256, 256, 0, stream>>>(qkvb, xtemp + l * 8, ob);
        gemm(ob, 384, xproj_w + (long long)l * 384 * 384, xproj_b + l * 384,
             tb, 384, xg1 + l * 384, tb, 384, 6272, 384, 384, 0, stream);
        ln_kernel<<<6272, 192, 0, stream>>>(tb, 384, hb, 384, xn3_w + l * 384, xn3_b + l * 384);
        lpi_kernel<<<32 * 384, 256, 0, stream>>>(hb,
            xlpi1_w + (long long)l * 384 * 9, xlpi1_b + l * 384,
            xbn_g + l * 384, xbn_b + l * 384,
            xlpi2_w + (long long)l * 384 * 9, xlpi2_b + l * 384,
            xg3 + l * 384, tb);
        ln_kernel<<<6272, 192, 0, stream>>>(tb, 384, hb, 384, xn2_w + l * 384, xn2_b + l * 384);
        gemm(hb, 384, xfc1_w + (long long)l * 1536 * 384, xfc1_b + l * 1536,
             hidb, 1536, nullptr, nullptr, 0, 6272, 1536, 384, 1, stream);
        gemm(hidb, 1536, xfc2_w + (long long)l * 384 * 1536, xfc2_b + l * 384,
             tb, 384, xg2 + l * 384, tb, 384, 6272, 384, 1536, 0, stream);
    }

    // ---- class-attention blocks ----
    float* t2   = ws + OFF_T2;
    float* clsb = ws + OFF_CLSB;
    float* clsp = ws + OFF_CLSP;
    float* cls2 = ws + OFF_CLS2;
    float* clsh = ws + OFF_CLSH;
    build_t2_kernel<<<(32 * 197 * 384 + 255) / 256, 256, 0, stream>>>(tb, cls_token, t2);
    for (int l = 0; l < 2; ++l) {
        ln_kernel<<<6304, 192, 0, stream>>>(t2, 384, hb, 384, cn1_w + l * 384, cn1_b + l * 384);
        gemm(hb, 384, cqkv_w + (long long)l * 1152 * 384, cqkv_b + l * 1152,
             qkvb, 1152, nullptr, nullptr, 0, 6304, 1152, 384, 0, stream);
        ca_attn_kernel<<<256, 256, 0, stream>>>(qkvb, clsb);
        gemm(clsb, 384, cproj_w + (long long)l * 384 * 384, cproj_b + l * 384,
             clsp, 384, nullptr, nullptr, 0, 32, 384, 384, 0, stream);
        ca_merge1_kernel<<<(32 * 197 * 384 + 255) / 256, 256, 0, stream>>>(t2, hb, clsp, cg1 + l * 384);
        ln_kernel<<<6304, 192, 0, stream>>>(t2, 384, t2, 384, cn2_w + l * 384, cn2_b + l * 384);
        gemm(t2, 197 * 384, cfc1_w + (long long)l * 1536 * 384, cfc1_b + l * 1536,
             clsh, 1536, nullptr, nullptr, 0, 32, 1536, 384, 1, stream);
        gemm(clsh, 1536, cfc2_w + (long long)l * 384 * 1536, cfc2_b + l * 384,
             cls2, 384, nullptr, nullptr, 0, 32, 384, 1536, 0, stream);
        ca_merge2_kernel<<<(32 * 197 * 384 + 255) / 256, 256, 0, stream>>>(t2, cls2, cg2 + l * 384);
    }

    // ---- final norm (token 0 only) + head ----
    ln_kernel<<<32, 192, 0, stream>>>(t2, 197 * 384, clsb, 384, norm_w, norm_b);
    gemm(clsb, 384, head_w, head_b, (float*)d_out, 1000,
         nullptr, nullptr, 0, 32, 1000, 384, 0, stream);
}
